// Model_34050500723161
// MI455X (gfx1250) — matrix operand order repaired and checked by reading, not compiled, not run on hardware
//
#include <hip/hip_runtime.h>

// ---------------------------------------------------------------------------
// Problem constants (match reference)
// ---------------------------------------------------------------------------
#define N_NODES   65536
#define N_EDGES   524288
#define N_GRAPHS  512
#define G_NODES   10000
#define NUM_DIS   2000
#define IN_DIM    128
#define HID       256
#define D_FEAT    512

// ---------------------------------------------------------------------------
// Types for WMMA
// ---------------------------------------------------------------------------
typedef __attribute__((ext_vector_type(16))) __bf16 bf16x16;
typedef __attribute__((ext_vector_type(8)))  __bf16 bf16x8;
typedef __attribute__((ext_vector_type(4)))  __bf16 bf16x4;
typedef __attribute__((ext_vector_type(8)))  float  f32x8;
typedef __attribute__((ext_vector_type(4)))  float  f32x4;

#define BM 128
#define BN 64

// ---------------------------------------------------------------------------
// Weight pre-transform: Wt[n][k] = bf16(W[k][n])   (col-major bf16 weights)
// This puts B fragments in ISA layout: two runs of 8 contiguous bf16 per lane.
// ---------------------------------------------------------------------------
__global__ __launch_bounds__(256)
void convert_weight_kernel(const float* __restrict__ W, __bf16* __restrict__ Wt,
                           int K, int N)
{
  const long long i = (long long)blockIdx.x * 256 + threadIdx.x;
  if (i >= (long long)K * N) return;
  const int n = (int)(i / K);
  const int k = (int)(i % K);
  Wt[i] = (__bf16)W[(long long)k * N + n];
}

// ---------------------------------------------------------------------------
// LDS-free WMMA GEMM:  C[M,N] = (relu?)( A[M,K] @ Bt^T (+bias) (+C) )
//   A:  f32 row-major [M][K]      (fragments loaded direct + v_cvt_pk_bf16)
//   Bt: bf16 col-major [N][K]     (fragments = contiguous global_load_b128 x2)
//   block = 256 threads = 8 waves; block tile 128x64; no LDS, no barriers.
//   wave w owns rows [16w,16w+16); 4 independent 16x16 accumulators (N dir).
// ---------------------------------------------------------------------------
template<bool RELU, bool BIAS, bool ACCUM>
__global__ __launch_bounds__(256)
void gemm_wmma_bf16(const float* __restrict__ A, const __bf16* __restrict__ Bt,
                    const float* __restrict__ bias, float* __restrict__ C,
                    int M, int N, int K)
{
  const int tid  = threadIdx.x;
  const int wave = tid >> 5;
  const int lane = tid & 31;
  const int lr   = lane & 15;
  const int half = lane >> 4;

  const int rowBase = blockIdx.x * BM;
  const int colBase = blockIdx.y * BN;
  const int row     = rowBase + wave * 16 + lr;   // A-fragment row for this lane

  f32x8 acc[4];
#pragma unroll
  for (int j = 0; j < 4; ++j)
#pragma unroll
    for (int r = 0; r < 8; ++r) acc[j][r] = 0.0f;

  const float*  arow = A + (long long)row * K;
  const __bf16* bcol[4];
#pragma unroll
  for (int j = 0; j < 4; ++j)
    bcol[j] = Bt + (long long)(colBase + j * 16 + lr) * K + half * 8;

  for (int kk = 0; kk < K; kk += 32) {
    // ---- A fragment (ISA 16-bit A 16x32 layout) ----
    // lane<16:  K = kk + {0..7, 16..23};  lane>=16: K = kk + {8..15, 24..31}
    bf16x16 a;
    if (row < M) {
      const float* ap = arow + kk + half * 8;
      f32x4 a0 = *(const f32x4*)(ap);
      f32x4 a1 = *(const f32x4*)(ap + 4);
      f32x4 a2 = *(const f32x4*)(ap + 16);
      f32x4 a3 = *(const f32x4*)(ap + 20);
      bf16x4 h0 = __builtin_convertvector(a0, bf16x4);
      bf16x4 h1 = __builtin_convertvector(a1, bf16x4);
      bf16x4 h2 = __builtin_convertvector(a2, bf16x4);
      bf16x4 h3 = __builtin_convertvector(a3, bf16x4);
      bf16x8 lo = __builtin_shufflevector(h0, h1, 0,1,2,3,4,5,6,7);
      bf16x8 hi = __builtin_shufflevector(h2, h3, 0,1,2,3,4,5,6,7);
      a = __builtin_shufflevector(lo, hi, 0,1,2,3,4,5,6,7,8,9,10,11,12,13,14,15);
    } else {
      f32x4 fz = {0.0f, 0.0f, 0.0f, 0.0f};
      bf16x4 hz = __builtin_convertvector(fz, bf16x4);
      bf16x8 z8 = __builtin_shufflevector(hz, hz, 0,1,2,3,0,1,2,3);
      a = __builtin_shufflevector(z8, z8, 0,1,2,3,4,5,6,7,8,9,10,11,12,13,14,15);
    }

#pragma unroll
    for (int j = 0; j < 4; ++j) {
      // B fragment: lane<16 -> col j*16+lr, K kk+{0..7,16..23} ; lane>=16 -> K kk+{8..15,24..31}
      const __bf16* bp = bcol[j] + kk;
      bf16x8 b0 = *(const bf16x8*)(bp);
      bf16x8 b1 = *(const bf16x8*)(bp + 16);
      bf16x16 b = __builtin_shufflevector(b0, b1,
          0,1,2,3,4,5,6,7,8,9,10,11,12,13,14,15);
      acc[j] = __builtin_amdgcn_wmma_f32_16x16x32_bf16(
          false, a, false, b, (short)0, acc[j], false, false);
    }
  }

  // ---- epilogue: C/D layout VGPR r -> row (half*8 + r), col = lr ----
#pragma unroll
  for (int j = 0; j < 4; ++j) {
    const int col = colBase + j * 16 + lr;
    const float bval = BIAS ? bias[col] : 0.0f;
#pragma unroll
    for (int r = 0; r < 8; ++r) {
      const int orow = rowBase + wave * 16 + half * 8 + r;
      if (orow < M) {
        const long long o = (long long)orow * N + col;
        float v = acc[j][r] + bval;
        if (ACCUM) v += C[o];
        if (RELU)  v = fmaxf(v, 0.0f);
        C[o] = v;
      }
    }
  }
}

// ---------------------------------------------------------------------------
// x[n] = concat( z_table[z[n]], node_id[n]<=2000 ? Pd[node_id] : Pc[node_id] )
// one thread per float4 chunk (64 chunks per node)
// ---------------------------------------------------------------------------
__global__ __launch_bounds__(256)
void build_x_kernel(const int* __restrict__ z, const int* __restrict__ nid,
                    const float* __restrict__ z_table,
                    const float* __restrict__ Pd, const float* __restrict__ Pc,
                    float* __restrict__ x)
{
  const long long idx = (long long)blockIdx.x * 256 + threadIdx.x; // N_NODES*64
  const int n = (int)(idx >> 6);
  const int q = (int)(idx & 63);
  float4 v;
  if (q < 32) {
    v = ((const float4*)(z_table + (long long)z[n] * IN_DIM))[q];
  } else {
    const int id = nid[n];
    const float* tab = (id <= NUM_DIS) ? Pd : Pc;
    v = ((const float4*)(tab + (long long)id * IN_DIM))[q - 32];
  }
  ((float4*)(x + (long long)n * HID))[q] = v;
}

// ---------------------------------------------------------------------------
// scatter:  agg[dst[e]] += feat[src[e]]   (+ deg[dst[e]] += 1, optional)
// 4 edges per 256-thread block; each lane moves a float4 (b128 load + 4 atomics)
// ---------------------------------------------------------------------------
__global__ __launch_bounds__(256)
void scatter_kernel(const int* __restrict__ src, const int* __restrict__ dst,
                    const float* __restrict__ feat, float* __restrict__ agg,
                    float* __restrict__ deg)
{
  const int t = threadIdx.x;
  const int e = blockIdx.x * 4 + (t >> 6);   // 4 edges / block
  const int l = t & 63;                      // 64 float4 lanes per edge
  const int s = src[e];
  const int d = dst[e];
  const float4 v = ((const float4*)(feat + (long long)s * HID))[l];
  float* base = agg + (long long)d * HID + l * 4;
  atomicAdd(base + 0, v.x);
  atomicAdd(base + 1, v.y);
  atomicAdd(base + 2, v.z);
  atomicAdd(base + 3, v.w);
  if (deg != nullptr && l == 0) atomicAdd(&deg[d], 1.0f);
}

__global__ __launch_bounds__(256)
void zero_kernel(float* __restrict__ p, long long n)
{
  long long i = (long long)blockIdx.x * 256 + threadIdx.x;
  const long long stride = (long long)gridDim.x * 256;
  for (; i < n; i += stride) p[i] = 0.0f;
}

__global__ __launch_bounds__(256)
void add_inplace_kernel(float* __restrict__ a, const float* __restrict__ b)
{
  const long long i = (long long)blockIdx.x * 256 + threadIdx.x;
  a[i] += b[i];
}

// agg[n,c] /= max(deg[n],1)
__global__ __launch_bounds__(256)
void scale_rows_kernel(float* __restrict__ agg, const float* __restrict__ deg)
{
  const long long i = (long long)blockIdx.x * 256 + threadIdx.x;
  const int n = (int)(i >> 8);           // HID == 256
  agg[i] *= 1.0f / fmaxf(deg[n], 1.0f);
}

// ---------------------------------------------------------------------------
// per-graph max pooling over sorted graph_ids (binary search for segment)
// one block per graph; thread c reduces column c
// ---------------------------------------------------------------------------
__global__ __launch_bounds__(256)
void segmax_kernel(const float* __restrict__ s, const int* __restrict__ gids,
                   float* __restrict__ g_out)
{
  const int g = blockIdx.x;
  int lo = 0, hi = N_NODES;
  while (lo < hi) { int mid = (lo + hi) >> 1; if (gids[mid] < g) lo = mid + 1; else hi = mid; }
  const int start = lo;
  hi = N_NODES;
  while (lo < hi) { int mid = (lo + hi) >> 1; if (gids[mid] <= g) lo = mid + 1; else hi = mid; }
  const int end = lo;

  const int c = threadIdx.x;
  float m = -__builtin_inff();
  for (int n = start; n < end; ++n)
    m = fmaxf(m, s[(long long)n * HID + c]);
  g_out[(long long)g * HID + c] = (start < end) ? m : 0.0f;  // isfinite guard
}

// out[g] = dot(h3[g,:], lin2_W) + lin2_b
__global__ __launch_bounds__(256)
void head_kernel(const float* __restrict__ h3, const float* __restrict__ W,
                 const float* __restrict__ b, float* __restrict__ out)
{
  __shared__ float red[256];
  const int g = blockIdx.x;
  const int t = threadIdx.x;
  red[t] = h3[(long long)g * HID + t] * W[t];
  __syncthreads();
#pragma unroll
  for (int s = 128; s > 0; s >>= 1) {
    if (t < s) red[t] += red[t + s];
    __syncthreads();
  }
  if (t == 0) out[g] = red[0] + b[0];
}

// ---------------------------------------------------------------------------
// launch
// ---------------------------------------------------------------------------
extern "C" void kernel_launch(void* const* d_in, const int* in_sizes, int n_in,
                              void* d_out, int out_size, void* d_ws, size_t ws_size,
                              hipStream_t stream)
{
  const int*   z          = (const int*)d_in[0];
  const int*   node_id    = (const int*)d_in[1];
  const int*   src        = (const int*)d_in[2];
  const int*   dst        = (const int*)d_in[3];
  const int*   gids       = (const int*)d_in[4];
  const float* z_table    = (const float*)d_in[5];
  const float* d_feat     = (const float*)d_in[6];
  const float* c_feat     = (const float*)d_in[7];
  const float* Wd         = (const float*)d_in[8];
  const float* Wc         = (const float*)d_in[9];
  const float* gin_W1     = (const float*)d_in[10];
  const float* gin_b1     = (const float*)d_in[11];
  const float* gin_W2     = (const float*)d_in[12];
  const float* gin_b2     = (const float*)d_in[13];
  const float* sage_Wself = (const float*)d_in[14];
  const float* sage_Wng   = (const float*)d_in[15];
  const float* sage_b     = (const float*)d_in[16];
  const float* lin1_W     = (const float*)d_in[17];
  const float* lin1_b     = (const float*)d_in[18];
  const float* lin2_W     = (const float*)d_in[19];
  const float* lin2_b     = (const float*)d_in[20];
  float* out = (float*)d_out;

  // workspace carve-up (floats first, then bf16 weight mirrors)
  float* ws   = (float*)d_ws;
  float* Pd   = ws; ws += (long long)G_NODES * IN_DIM;   // projected disease table
  float* Pc   = ws; ws += (long long)G_NODES * IN_DIM;   // projected chemical table
  float* x    = ws; ws += (long long)N_NODES * HID;      // x, then t1 = x+agg (in place)
  float* agg  = ws; ws += (long long)N_NODES * HID;      // GIN agg, then SAGE mean
  float* t2   = ws; ws += (long long)N_NODES * HID;      // MLP hidden, then SAGE out
  float* h    = ws; ws += (long long)N_NODES * HID;      // GIN output
  float* deg  = ws; ws += N_NODES;
  float* gbuf = ws; ws += (long long)N_GRAPHS * HID;
  float* h3   = ws; ws += (long long)N_GRAPHS * HID;

  __bf16* bw    = (__bf16*)ws;
  __bf16* WdT   = bw; bw += (long long)D_FEAT * IN_DIM;  // [IN_DIM][D_FEAT]
  __bf16* WcT   = bw; bw += (long long)D_FEAT * IN_DIM;
  __bf16* W1T   = bw; bw += (long long)HID * HID;        // [HID][HID]
  __bf16* W2T   = bw; bw += (long long)HID * HID;
  __bf16* WsT   = bw; bw += (long long)HID * HID;
  __bf16* WnT   = bw; bw += (long long)HID * HID;
  __bf16* L1T   = bw; bw += (long long)HID * HID;

  const dim3 blk(256);
  const dim3 gProj((G_NODES + BM - 1) / BM, IN_DIM / BN);
  const dim3 gNode(N_NODES / BM, HID / BN);
  const dim3 gGraph((N_GRAPHS + BM - 1) / BM, HID / BN);
  const int  cwProj = (D_FEAT * IN_DIM + 255) / 256;
  const int  cwHid  = (HID * HID + 255) / 256;

  // 0) pre-transpose + convert weights to bf16 col-major (tiny, done per call)
  convert_weight_kernel<<<cwProj, blk, 0, stream>>>(Wd, WdT, D_FEAT, IN_DIM);
  convert_weight_kernel<<<cwProj, blk, 0, stream>>>(Wc, WcT, D_FEAT, IN_DIM);
  convert_weight_kernel<<<cwHid,  blk, 0, stream>>>(gin_W1, W1T, HID, HID);
  convert_weight_kernel<<<cwHid,  blk, 0, stream>>>(gin_W2, W2T, HID, HID);
  convert_weight_kernel<<<cwHid,  blk, 0, stream>>>(sage_Wself, WsT, HID, HID);
  convert_weight_kernel<<<cwHid,  blk, 0, stream>>>(sage_Wng,  WnT, HID, HID);
  convert_weight_kernel<<<cwHid,  blk, 0, stream>>>(lin1_W, L1T, HID, HID);

  // 1) project the 10000-row tables once (6.5x fewer FLOPs than per-node proj)
  gemm_wmma_bf16<false,false,false><<<gProj, blk, 0, stream>>>(d_feat, WdT, nullptr, Pd, G_NODES, IN_DIM, D_FEAT);
  gemm_wmma_bf16<false,false,false><<<gProj, blk, 0, stream>>>(c_feat, WcT, nullptr, Pc, G_NODES, IN_DIM, D_FEAT);

  // 2) x = concat(z_emb, select(d/c proj))
  build_x_kernel<<<(N_NODES * 64) / 256, blk, 0, stream>>>(z, node_id, z_table, Pd, Pc, x);

  // 3) GIN aggregation: agg = segment_sum(x[src] -> dst), deg = in-degree
  zero_kernel<<<4096, blk, 0, stream>>>(agg, (long long)N_NODES * HID);
  zero_kernel<<<64,   blk, 0, stream>>>(deg, N_NODES);
  scatter_kernel<<<N_EDGES / 4, blk, 0, stream>>>(src, dst, x, agg, deg);
  add_inplace_kernel<<<(N_NODES * HID) / 256, blk, 0, stream>>>(x, agg);   // t1 = x + agg

  // 4) GIN MLP: h = relu(t1@W1+b1)@W2+b2
  gemm_wmma_bf16<true, true, false><<<gNode, blk, 0, stream>>>(x,  W1T, gin_b1, t2, N_NODES, HID, HID);
  gemm_wmma_bf16<false,true, false><<<gNode, blk, 0, stream>>>(t2, W2T, gin_b2, h,  N_NODES, HID, HID);

  // 5) SAGE mean aggregation of h
  zero_kernel<<<4096, blk, 0, stream>>>(agg, (long long)N_NODES * HID);
  scatter_kernel<<<N_EDGES / 4, blk, 0, stream>>>(src, dst, h, agg, nullptr);
  scale_rows_kernel<<<(N_NODES * HID) / 256, blk, 0, stream>>>(agg, deg);

  // 6) SAGE: s = h@Wself + b  (into t2), then s += mean@Wneigh
  gemm_wmma_bf16<false,true, false><<<gNode, blk, 0, stream>>>(h,   WsT, sage_b, t2, N_NODES, HID, HID);
  gemm_wmma_bf16<false,false,true ><<<gNode, blk, 0, stream>>>(agg, WnT, nullptr, t2, N_NODES, HID, HID);

  // 7) per-graph max pooling (sorted graph_ids, no atomics)
  segmax_kernel<<<N_GRAPHS, blk, 0, stream>>>(t2, gids, gbuf);

  // 8) head: h3 = relu(g@lin1+b1); out = h3@lin2 + b2
  gemm_wmma_bf16<true, true, false><<<gGraph, blk, 0, stream>>>(gbuf, L1T, lin1_b, h3, N_GRAPHS, HID, HID);
  head_kernel<<<N_GRAPHS, blk, 0, stream>>>(h3, lin2_W, lin2_b, out);
}
